// simple_RGIN_model_77163382440873
// MI455X (gfx1250) — matrix operand order repaired and checked by reading, not compiled, not run on hardware
//
#include <hip/hip_runtime.h>
#include <math.h>

// ---------------------------------------------------------------------------
// CDNA5 / gfx1250 types for WMMA (wave32)
// ---------------------------------------------------------------------------
typedef __attribute__((ext_vector_type(16))) __bf16        v16bf;
typedef __attribute__((ext_vector_type(8)))  float         v8f;
typedef __attribute__((ext_vector_type(8)))  unsigned int  v8u;
typedef __attribute__((ext_vector_type(4)))  unsigned int  v4u;

__device__ __forceinline__ unsigned short f32_to_bf16(float f) {
    unsigned int u = __float_as_uint(f);
    u += 0x7FFFu + ((u >> 16) & 1u);   // round-to-nearest-even
    return (unsigned short)(u >> 16);
}

#define DD 128          // feature dim (D)
#define BN_EPS 1e-3f
#define L2_EPS 1e-12f

// ---------------------------------------------------------------------------
// K0: transpose + cast W1, Wd (D x D fp32, row-major) -> column-major bf16
//     so each WMMA B-fragment is a contiguous 32B load.
// ---------------------------------------------------------------------------
__global__ void castW_kernel(const float* __restrict__ W1,
                             const float* __restrict__ Wd,
                             unsigned short* __restrict__ Wt1,
                             unsigned short* __restrict__ Wtd) {
    int t = blockIdx.x * blockDim.x + threadIdx.x;
    if (t >= DD * DD) return;
    int n = t & (DD - 1);
    int k = t >> 7;
    Wt1[n * DD + k] = f32_to_bf16(W1[k * DD + n]);
    Wtd[n * DD + k] = f32_to_bf16(Wd[k * DD + n]);
}

// ---------------------------------------------------------------------------
// K1: h16 = bf16(BN1(x)) ; acc = 0  (acc is the SpMM-1 accumulation buffer)
// ---------------------------------------------------------------------------
__global__ void bn1_cast_kernel(const float* __restrict__ x,
                                const float* __restrict__ g,
                                const float* __restrict__ b,
                                const float* __restrict__ mu,
                                const float* __restrict__ var,
                                unsigned short* __restrict__ h16,
                                float* __restrict__ acc,
                                int total) {
    int i = blockIdx.x * blockDim.x + threadIdx.x;
    if (i >= total) return;
    int d = i & (DD - 1);
    float v = (x[i] - mu[d]) * rsqrtf(var[d] + BN_EPS) * g[d] + b[d];
    h16[i] = f32_to_bf16(v);
    acc[i] = 0.0f;
}

// ---------------------------------------------------------------------------
// GEMM: C[M x 128] = A16[M x 128] @ Bt16(col-major 128x128)  (+ bias)
// One wave -> one 16x16 C tile; 8 waves/block cover all 8 N-tiles of a
// 16-row stripe. 4 x v_wmma_f32_16x16x32_bf16 per tile (K = 128).
//
// ISA 16-bit A layout (16x32): lanes 0-15 hold row M=lane, K = {0..7,16..23};
// lanes 16-31 hold row M=lane-16, K = {8..15,24..31}. This code instead loads A and B alike: lanes 0-15
// hold row M / col N = lane, K = 0..15 contiguous; lanes 16-31 K = 16..31.
// ---------------------------------------------------------------------------
__global__ __launch_bounds__(256) void gemm_bf16_kernel(
        const unsigned short* __restrict__ A16,
        const unsigned short* __restrict__ Bt16,
        const float* __restrict__ bias,
        float* __restrict__ C,
        int hasBias) {
    const int lane  = threadIdx.x & 31;
    const int wave  = threadIdx.x >> 5;
    const int tileM = blockIdx.x << 4;
    const int tileN = wave << 4;
    const int lmod  = lane & 15;
    const int hi    = lane >> 4;          // 0 for lanes 0-15, 1 for 16-31

    v8f acc = {};

    const unsigned short* arow = A16 + (size_t)(tileM + lmod) * DD + (hi ? 16 : 0);
    const unsigned short* bcol = Bt16 + (size_t)(tileN + lmod) * DD + (hi ? 16 : 0);

#pragma unroll
    for (int kk = 0; kk < 4; ++kk) {
        const int kb = kk * 32;
        v4u a0 = *(const v4u*)(arow + kb);        // K base+0 .. +7   (16B)
        v4u a1 = *(const v4u*)(arow + kb + 8);    // K base+8 .. +15  (16B)
        v8u av;
        av[0] = a0[0]; av[1] = a0[1]; av[2] = a0[2]; av[3] = a0[3];
        av[4] = a1[0]; av[5] = a1[1]; av[6] = a1[2]; av[7] = a1[3];
        v8u bv = *(const v8u*)(bcol + kb);        // 16 contiguous bf16 (32B)

        v16bf a = __builtin_bit_cast(v16bf, av);
        v16bf b = __builtin_bit_cast(v16bf, bv);
        acc = __builtin_amdgcn_wmma_f32_16x16x32_bf16(
                false, a, false, b, (short)0, acc, false, false);
    }

    // C/D 16x16 f32 layout: lane<16 -> N=lane, M=r; lane>=16 -> N=lane-16, M=r+8
    const int cn = tileN + lmod;
    const float bb = hasBias ? bias[cn] : 0.0f;
    float* crow = C + (size_t)(tileM + (hi ? 8 : 0)) * DD + cn;
#pragma unroll
    for (int r = 0; r < 8; ++r)
        crow[(size_t)r * DD] = acc[r] + bb;
}

// ---------------------------------------------------------------------------
// SpMM scatter: dst[rows[e]] += w_e * src[cols[e]]  (one wave per edge,
// float4 gather from L2-resident src, 4 global_atomic_add_f32 per lane).
// ---------------------------------------------------------------------------
__global__ __launch_bounds__(256) void spmm1_kernel(
        const int* __restrict__ rows, const int* __restrict__ cols,
        const float* __restrict__ vals,
        const float* __restrict__ src, float* __restrict__ dst, int E) {
    int e = blockIdx.x * 8 + (threadIdx.x >> 5);
    if (e >= E) return;
    int lane = threadIdx.x & 31;
    int r = rows[e], c = cols[e];
    float w = vals[e];
    const float4 s = *(const float4*)(src + (size_t)c * DD + lane * 4);
    float* d = dst + (size_t)r * DD + lane * 4;
    atomicAdd(d + 0, w * s.x);
    atomicAdd(d + 1, w * s.y);
    atomicAdd(d + 2, w * s.z);
    atomicAdd(d + 3, w * s.w);
}

__global__ __launch_bounds__(256) void spmm2_kernel(
        const int* __restrict__ rows, const int* __restrict__ cols,
        const float* __restrict__ vals, const int* __restrict__ rel,
        const float* __restrict__ relc,
        const float* __restrict__ src, float* __restrict__ dst, int E) {
    int e = blockIdx.x * 8 + (threadIdx.x >> 5);
    if (e >= E) return;
    int lane = threadIdx.x & 31;
    int r = rows[e], c = cols[e];
    float w = vals[e] / (relc[rel[e]] + 1.0f);
    const float4 s = *(const float4*)(src + (size_t)c * DD + lane * 4);
    float* d = dst + (size_t)r * DD + lane * 4;
    atomicAdd(d + 0, w * s.x);
    atomicAdd(d + 1, w * s.y);
    atomicAdd(d + 2, w * s.z);
    atomicAdd(d + 3, w * s.w);
}

// ---------------------------------------------------------------------------
// K5 (fused): y1 = tanh(acc + b1); h2 = BN2(y1); agg = 0
// ---------------------------------------------------------------------------
__global__ void post1_kernel(float* __restrict__ y1,
                             const float* __restrict__ b1,
                             const float* __restrict__ g2,
                             const float* __restrict__ be2,
                             const float* __restrict__ mu2,
                             const float* __restrict__ var2,
                             float* __restrict__ h2,
                             float* __restrict__ agg,
                             int total) {
    int i = blockIdx.x * blockDim.x + threadIdx.x;
    if (i >= total) return;
    int d = i & (DD - 1);
    float v = tanhf(y1[i] + b1[d]);
    y1[i] = v;
    h2[i] = (v - mu2[d]) * rsqrtf(var2[d] + BN_EPS) * g2[d] + be2[d];
    agg[i] = 0.0f;
}

// ---------------------------------------------------------------------------
// K7: a16 = bf16(agg + h2 * (coeff_kernel[row] + 1))
// ---------------------------------------------------------------------------
__global__ void selfcast_kernel(const float* __restrict__ agg,
                                const float* __restrict__ h2,
                                const float* __restrict__ ck,
                                unsigned short* __restrict__ a16,
                                int total) {
    int i = blockIdx.x * blockDim.x + threadIdx.x;
    if (i >= total) return;
    int row = i >> 7;
    a16[i] = f32_to_bf16(agg[i] + h2[i] * (ck[row] + 1.0f));
}

// ---------------------------------------------------------------------------
// K9: out[row] = l2n(concat(l2n(x), l2n(y1), l2n(y2)))   (one wave per row)
// ---------------------------------------------------------------------------
__global__ __launch_bounds__(256) void finalize_kernel(
        const float* __restrict__ x,
        const float* __restrict__ y1,
        const float* __restrict__ y2,
        float* __restrict__ out, int Nn) {
    int row  = blockIdx.x * 8 + (threadIdx.x >> 5);
    if (row >= Nn) return;
    int lane = threadIdx.x & 31;

    const float4 xv = *(const float4*)(x  + (size_t)row * DD + lane * 4);
    const float4 v1 = *(const float4*)(y1 + (size_t)row * DD + lane * 4);
    const float4 v2 = *(const float4*)(y2 + (size_t)row * DD + lane * 4);

    float sx = xv.x * xv.x + xv.y * xv.y + xv.z * xv.z + xv.w * xv.w;
    float s1 = v1.x * v1.x + v1.y * v1.y + v1.z * v1.z + v1.w * v1.w;
    float s2 = v2.x * v2.x + v2.y * v2.y + v2.z * v2.z + v2.w * v2.w;

#pragma unroll
    for (int m = 16; m > 0; m >>= 1) {
        sx += __shfl_xor(sx, m, 32);
        s1 += __shfl_xor(s1, m, 32);
        s2 += __shfl_xor(s2, m, 32);
    }
    float rsx = rsqrtf(fmaxf(sx, L2_EPS));
    float rs1 = rsqrtf(fmaxf(s1, L2_EPS));
    float rs2 = rsqrtf(fmaxf(s2, L2_EPS));
    // squared norm of the concatenation of the three normalized pieces
    float tot  = sx * rsx * rsx + s1 * rs1 * rs1 + s2 * rs2 * rs2;
    float rall = rsqrtf(fmaxf(tot, L2_EPS));

    float fx = rsx * rall, f1 = rs1 * rall, f2 = rs2 * rall;
    float* o = out + (size_t)row * (3 * DD) + lane * 4;
    *(float4*)(o +   0) = make_float4(xv.x * fx, xv.y * fx, xv.z * fx, xv.w * fx);
    *(float4*)(o + 128) = make_float4(v1.x * f1, v1.y * f1, v1.z * f1, v1.w * f1);
    *(float4*)(o + 256) = make_float4(v2.x * f2, v2.y * f2, v2.z * f2, v2.w * f2);
}

// ---------------------------------------------------------------------------
// Host-side launch
// ---------------------------------------------------------------------------
extern "C" void kernel_launch(void* const* d_in, const int* in_sizes, int n_in,
                              void* d_out, int out_size, void* d_ws, size_t ws_size,
                              hipStream_t stream) {
    const float* x    = (const float*)d_in[0];
    const int*   rows = (const int*)  d_in[1];
    const int*   cols = (const int*)  d_in[2];
    const float* adj  = (const float*)d_in[3];
    const int*   rel  = (const int*)  d_in[4];
    const float* g1   = (const float*)d_in[5];
    const float* be1  = (const float*)d_in[6];
    const float* mu1  = (const float*)d_in[7];
    const float* var1 = (const float*)d_in[8];
    const float* W1   = (const float*)d_in[9];
    const float* b1   = (const float*)d_in[10];
    const float* g2   = (const float*)d_in[11];
    const float* be2  = (const float*)d_in[12];
    const float* mu2  = (const float*)d_in[13];
    const float* var2 = (const float*)d_in[14];
    const float* relc = (const float*)d_in[15];
    const float* ck   = (const float*)d_in[16];
    const float* Wd   = (const float*)d_in[17];
    const float* bd   = (const float*)d_in[18];
    float* out = (float*)d_out;

    const int N = in_sizes[0] / DD;
    const int E = in_sizes[1];
    const int total = N * DD;

    // Workspace layout (base assumed >=256B aligned):
    //   bufA16 : N*128 bf16  (BN1(x) bf16; later reused as GEMM-2 A)
    //   bufF1  : N*128 f32   (hW; later reused as SpMM-2 accumulator)
    //   bufF2  : N*128 f32   (y1, kept to the end)
    //   bufF3  : N*128 f32   (h2; later reused as y2)
    //   Wt1,Wtd: 128*128 bf16 each
    char* ws = (char*)d_ws;
    size_t szA16 = (size_t)total * sizeof(unsigned short);
    size_t szF   = (size_t)total * sizeof(float);
    unsigned short* bufA16 = (unsigned short*)ws;
    float* bufF1 = (float*)(ws + szA16);
    float* bufF2 = (float*)(ws + szA16 + szF);
    float* bufF3 = (float*)(ws + szA16 + 2 * szF);
    unsigned short* Wt1 = (unsigned short*)(ws + szA16 + 3 * szF);
    unsigned short* Wtd = Wt1 + DD * DD;

    const int ethreads = 256;
    const int eblocks  = (total + ethreads - 1) / ethreads;
    const int mtiles   = N / 16;                 // N = 50000 -> 3125 exact
    const int edgeblk  = (E + 7) / 8;            // 8 waves (edges) per block
    const int rowblk   = (N + 7) / 8;            // 8 rows per block

    // K0: weight transpose+cast
    castW_kernel<<<(DD * DD + 255) / 256, 256, 0, stream>>>(W1, Wd, Wt1, Wtd);
    // K1: BN1 + cast, zero SpMM-1 accumulator (bufF2)
    bn1_cast_kernel<<<eblocks, 256, 0, stream>>>(x, g1, be1, mu1, var1,
                                                 bufA16, bufF2, total);
    // K2: hW = BN1(x) @ W1   (WMMA)
    gemm_bf16_kernel<<<mtiles, 256, 0, stream>>>(bufA16, Wt1, nullptr, bufF1, 0);
    // K4: SpMM-1 scatter into bufF2
    spmm1_kernel<<<edgeblk, 256, 0, stream>>>(rows, cols, adj, bufF1, bufF2, E);
    // K5: y1 = tanh(+b1); h2 = BN2(y1); zero SpMM-2 accumulator (bufF1)
    post1_kernel<<<eblocks, 256, 0, stream>>>(bufF2, b1, g2, be2, mu2, var2,
                                              bufF3, bufF1, total);
    // K6: SpMM-2 (relation-weighted) into bufF1
    spmm2_kernel<<<edgeblk, 256, 0, stream>>>(rows, cols, adj, rel, relc,
                                              bufF3, bufF1, E);
    // K7: self term + bf16 cast into bufA16
    selfcast_kernel<<<eblocks, 256, 0, stream>>>(bufF1, bufF3, ck, bufA16, total);
    // K8: y2 = (...) @ Wd + bd   (WMMA; writes bufF3, h2 dead)
    gemm_bf16_kernel<<<mtiles, 256, 0, stream>>>(bufA16, Wtd, bd, bufF3, 1);
    // K9: concat of per-piece l2-norms, renormalized
    finalize_kernel<<<rowblk, 256, 0, stream>>>(x, bufF2, bufF3, out, N);
}
